// SequenceParallelAttention_80281528697565
// MI455X (gfx1250) — matrix operand order repaired and checked by reading, compiled, not run on hardware
//
#include <hip/hip_runtime.h>

// ---------------------------------------------------------------------------
// Problem constants (match the reference)
// ---------------------------------------------------------------------------
#define SP_ 4
#define NH_ 16
#define HD_ 64
#define HS_ 1024
#define S_  8192
#define SC_ (S_ / SP_)          // 2048

typedef __attribute__((ext_vector_type(16))) __bf16 v16bf;
typedef __attribute__((ext_vector_type(8)))  __bf16 v8bf;
typedef __attribute__((ext_vector_type(8)))  float  v8f;

union BF16Frag { v16bf v; v8bf h[2]; };

static __device__ __forceinline__ v8f vzero8() {
    v8f z = {0.f, 0.f, 0.f, 0.f, 0.f, 0.f, 0.f, 0.f};
    return z;
}

// D = A(16x32 bf16) * B(32x16 bf16) + C(16x16 f32)
static __device__ __forceinline__ v8f wmma_bf16(const BF16Frag& a, const BF16Frag& b, v8f c) {
    return __builtin_amdgcn_wmma_f32_16x16x32_bf16(
        /*neg_a=*/false, a.v, /*neg_b=*/false, b.v,
        /*c_mod=*/(short)0, c, /*reuse_a=*/false, /*reuse_b=*/false);
}

// ---- CDNA5 async global->LDS copy (16B per lane), tracked by ASYNCcnt ------
static __device__ __forceinline__ void async_load_b128(void* lds_dst, const void* gsrc) {
    // Generic shared pointer: low 32 bits are the LDS byte address (ISA aperture rule).
    unsigned lds = (unsigned)(unsigned long long)lds_dst;
    unsigned long long ga = (unsigned long long)gsrc;
    asm volatile("global_load_async_to_lds_b128 %0, %1, off"
                 :: "v"(lds), "v"(ga)
                 : "memory");
}

static __device__ __forceinline__ void wait_async0() {
#if __has_builtin(__builtin_amdgcn_s_wait_asynccnt)
    __builtin_amdgcn_s_wait_asynccnt(0);
#else
    asm volatile("s_wait_asynccnt 0" ::: "memory");
#endif
}

// ---------------------------------------------------------------------------
// fp32 -> bf16 conversion (grid-stride)
// ---------------------------------------------------------------------------
__global__ void __launch_bounds__(256)
cvt_f32_bf16_kernel(const float* __restrict__ src, __bf16* __restrict__ dst, int n) {
    int i      = blockIdx.x * blockDim.x + threadIdx.x;
    int stride = gridDim.x * blockDim.x;
    for (; i < n; i += stride) dst[i] = (__bf16)src[i];
}

// ---------------------------------------------------------------------------
// bf16 GEMM:  C[M,N] = A[M,K] * W[K,N] + bias[N]
// Block tile 128x128, K-step 32. 256 threads = 8 waves (wave32).
// Wave grid 2(M) x 4(N): each wave owns 64x32 = 4x2 WMMA 16x16 tiles.
// Double-buffered LDS; A tiles prefetched with global_load_async_to_lds_b128
// so the async unit runs concurrently with the WMMA stream.
// ---------------------------------------------------------------------------
template <bool OUT_F32>
__global__ void __launch_bounds__(256)
gemm_bf16_kernel(const __bf16* __restrict__ A, const __bf16* __restrict__ W,
                 const float* __restrict__ bias, void* __restrict__ Cout,
                 int M, int N, int K) {
    constexpr int BM = 128, BN = 128, BK = 32;
    __shared__ __align__(16) __bf16 Ash[2][BM][BK];    // [buf][row][k]
    __shared__ __align__(16) __bf16 Bsh[2][BN][BK];    // [buf][n][k] = W[k][n0+n]

    const int t    = threadIdx.x;
    const int lane = t & 31;
    const int wave = t >> 5;
    const int wm   = wave & 1;          // 0..1 : 64-row slab
    const int wn   = wave >> 1;         // 0..3 : 32-col slab
    const int m0   = blockIdx.x * BM;
    const int n0   = blockIdx.y * BN;

    v8f acc[4][2];
#pragma unroll
    for (int i = 0; i < 4; ++i)
#pragma unroll
        for (int j = 0; j < 2; ++j) acc[i][j] = vzero8();

    const int lrow  = lane & 15;
    const int ahalf = (lane >> 4) * 8;   // A frag K-half base (0 / 8)
    const int bbase = (lane >> 4) * 8;   // B frag K-half base (0 / 8)

    auto stage = [&](int k0, int buf) {
        // A tile 128x32: 512 x 16B async segments, 2 per thread (ASYNCcnt)
#pragma unroll
        for (int it = 0; it < 2; ++it) {
            int idx = t + it * 256;            // 0..511
            int row = idx >> 2, seg = idx & 3;
            async_load_b128(&Ash[buf][row][seg * 8],
                            &A[(size_t)(m0 + row) * K + k0 + seg * 8]);
        }
        // B tile 32x128 transposed on the fly (coalesced reads, ds scatter)
#pragma unroll
        for (int it = 0; it < 2; ++it) {
            int idx = t + it * 256;            // 0..511 = 32(k) x 16(n-seg)
            int kk  = idx >> 4;
            int ns  = (idx & 15) * 8;
            v8bf wrow = *(const v8bf*)&W[(size_t)(k0 + kk) * N + n0 + ns];
#pragma unroll
            for (int j = 0; j < 8; ++j) Bsh[buf][ns + j][kk] = wrow[j];
        }
    };

    stage(0, 0);
    wait_async0();
    __syncthreads();

    int cur = 0;
    for (int k0 = 0; k0 < K; k0 += BK) {
        const int nxt = cur ^ 1;
        if (k0 + BK < K) stage(k0 + BK, nxt);   // overlaps with WMMAs below

        // ---- 8 WMMAs per wave on buffer `cur`
#pragma unroll
        for (int mt = 0; mt < 4; ++mt) {
            BF16Frag af;
            int arow = wm * 64 + mt * 16 + lrow;
            af.h[0] = *(const v8bf*)&Ash[cur][arow][ahalf];
            af.h[1] = *(const v8bf*)&Ash[cur][arow][ahalf + 16];
#pragma unroll
            for (int nt = 0; nt < 2; ++nt) {
                BF16Frag bf;
                int bcol = wn * 32 + nt * 16 + lrow;
                bf.h[0] = *(const v8bf*)&Bsh[cur][bcol][bbase];
                bf.h[1] = *(const v8bf*)&Bsh[cur][bcol][bbase + 16];
                acc[mt][nt] = wmma_bf16(af, bf, acc[mt][nt]);
            }
        }

        wait_async0();          // next tile's async writes complete (this wave)
        __syncthreads();        // all waves' writes visible, readers done
        cur = nxt;
    }

    // ---- epilogue: C layout -> row = (lane>>4)*8 + r ; col = lane&15 in tile
    const int rhalf = (lane >> 4) * 8;
#pragma unroll
    for (int mt = 0; mt < 4; ++mt) {
#pragma unroll
        for (int nt = 0; nt < 2; ++nt) {
#pragma unroll
            for (int r = 0; r < 8; ++r) {
                int row = m0 + wm * 64 + mt * 16 + rhalf + r;
                int col = n0 + wn * 32 + nt * 16 + lrow;
                float v = acc[mt][nt][r] + bias[col];
                if (OUT_F32) ((float*)Cout)[(size_t)row * N + col] = v;
                else         ((__bf16*)Cout)[(size_t)row * N + col] = (__bf16)v;
            }
        }
    }
}

// ---------------------------------------------------------------------------
// Ring attention, per-chunk softmax (flash-style online softmax inside each
// 2048-key chunk), out = (1/SP) * sum_chunks softmax(QK^T/sqrt(HD)) V.
// Grid: (SC/128, NH, SP). Block: 256 threads = 8 waves; wave w owns q-rows
// [w*16, w*16+16) of the 128-row query tile, full HD=64 output.
// K tiles staged with async global->LDS; V staged via VGPR transpose.
// ---------------------------------------------------------------------------
__global__ void __launch_bounds__(256)
ring_attention_kernel(const __bf16* __restrict__ Q, const __bf16* __restrict__ K,
                      const __bf16* __restrict__ V, __bf16* __restrict__ Ctx) {
    constexpr int QT = 128, KT = 64;
    __shared__ __align__(16) __bf16 Qsh[QT][HD_];      // pre-scaled by 1/sqrt(HD)
    __shared__ __align__(16) __bf16 Ksh[KT][HD_];      // [key][d]
    __shared__ __align__(16) __bf16 Vsh[HD_][KT];      // transposed: [d][key]
    __shared__ __align__(16) __bf16 Psh[8][16][KT];    // per-wave probs [qrow][key]

    const int t    = threadIdx.x;
    const int lane = t & 31;
    const int wave = t >> 5;
    const int qt   = blockIdx.x;
    const int h    = blockIdx.y;
    const int rch  = blockIdx.z;                 // this rank's chunk

    const int q_row0 = rch * SC_ + qt * QT;      // global query row base
    const int colh   = h * HD_;                  // head column offset in HS

    const int lrow  = lane & 15;
    const int half  = lane >> 4;                 // 0/1
    const int rhalf = half * 8;

    // ---- load + scale Q tile: 128x64, 1024 x 16B segments, 4 per thread
#pragma unroll
    for (int it = 0; it < 4; ++it) {
        int idx = t + it * 256;                  // 0..1023
        int row = idx >> 3, seg = idx & 7;
        v8bf qv = *(const v8bf*)&Q[(size_t)(q_row0 + row) * HS_ + colh + seg * 8];
#pragma unroll
        for (int j = 0; j < 8; ++j)
            Qsh[row][seg * 8 + j] = (__bf16)((float)qv[j] * 0.125f);  // 1/sqrt(64)
    }
    __syncthreads();

    // ---- per-wave constant Q A-fragments (d halves 0..31, 32..63)
    BF16Frag aq[2];
    {
        int qrow = wave * 16 + lrow;
#pragma unroll
        for (int dh = 0; dh < 2; ++dh) {
            aq[dh].h[0] = *(const v8bf*)&Qsh[qrow][dh * 32 + rhalf];        // K 0..7 / 8..15
            aq[dh].h[1] = *(const v8bf*)&Qsh[qrow][dh * 32 + rhalf + 16];   // K 16..23 / 24..31
        }
    }

    v8f out_total[4];
#pragma unroll
    for (int dt = 0; dt < 4; ++dt) out_total[dt] = vzero8();

    for (int step = 0; step < SP_; ++step) {
        const int c = (rch + step) & (SP_ - 1);         // ring: chunk (r+i) mod SP
        const int krow0 = c * SC_;

        float m[8], l[8];
        v8f acc_o[4];
#pragma unroll
        for (int r = 0; r < 8; ++r) { m[r] = -3.0e38f; l[r] = 0.f; }
#pragma unroll
        for (int dt = 0; dt < 4; ++dt) acc_o[dt] = vzero8();

        for (int kt = 0; kt < SC_ / KT; ++kt) {
            const int key0 = krow0 + kt * KT;
            __syncthreads();                      // prior users of Ksh/Vsh done
            // ---- stage K (async, ASYNCcnt) and V (transposed via VGPRs)
#pragma unroll
            for (int it = 0; it < 2; ++it) {
                int idx = t + it * 256;           // 0..511
                int row = idx >> 3, seg = idx & 7;
                async_load_b128(&Ksh[row][seg * 8],
                                &K[(size_t)(key0 + row) * HS_ + colh + seg * 8]);
                v8bf vv = *(const v8bf*)&V[(size_t)(key0 + row) * HS_ + colh + seg * 8];
#pragma unroll
                for (int j = 0; j < 8; ++j) Vsh[seg * 8 + j][row] = vv[j];
            }
            wait_async0();
            __syncthreads();

            // ---- S = Q K^T for 16q x 64k : 4 tiles x 2 WMMA (d halves)
            v8f acc_s[4];
#pragma unroll
            for (int nt = 0; nt < 4; ++nt) {
                v8f s = vzero8();
#pragma unroll
                for (int dh = 0; dh < 2; ++dh) {
                    // B = K^T (32d x 16key): lane col = key, d in table order (8-halves)
                    BF16Frag bk;
                    int kcol = nt * 16 + lrow;
                    int db   = dh * 32 + half * 8;
                    bk.h[0] = *(const v8bf*)&Ksh[kcol][db];
                    bk.h[1] = *(const v8bf*)&Ksh[kcol][db + 16];
                    s = wmma_bf16(aq[dh], bk, s);
                }
                acc_s[nt] = s;
            }

            // ---- online softmax: row max across the 64 keys of this tile
            float mt_[8];
#pragma unroll
            for (int r = 0; r < 8; ++r) {
                float v = acc_s[0][r];
                v = fmaxf(v, acc_s[1][r]);
                v = fmaxf(v, acc_s[2][r]);
                v = fmaxf(v, acc_s[3][r]);
                v = fmaxf(v, __shfl_xor(v, 1));
                v = fmaxf(v, __shfl_xor(v, 2));
                v = fmaxf(v, __shfl_xor(v, 4));
                v = fmaxf(v, __shfl_xor(v, 8));   // stays inside 16-lane half
                mt_[r] = v;
            }
#pragma unroll
            for (int r = 0; r < 8; ++r) {
                float nm = fmaxf(m[r], mt_[r]);
                float sc = __expf(m[r] - nm);
                m[r] = nm;
                l[r] *= sc;
#pragma unroll
                for (int dt = 0; dt < 4; ++dt) acc_o[dt][r] *= sc;
            }

            // ---- P = exp(S - m), stash to LDS in [qrow][key] for A-fragmenting
            float rs[8];
#pragma unroll
            for (int r = 0; r < 8; ++r) rs[r] = 0.f;
#pragma unroll
            for (int nt = 0; nt < 4; ++nt) {
#pragma unroll
                for (int r = 0; r < 8; ++r) {
                    float p = __expf(acc_s[nt][r] - m[r]);
                    rs[r] += p;
                    Psh[wave][rhalf + r][nt * 16 + lrow] = (__bf16)p;
                }
            }
#pragma unroll
            for (int r = 0; r < 8; ++r) {
                float v = rs[r];
                v += __shfl_xor(v, 1);
                v += __shfl_xor(v, 2);
                v += __shfl_xor(v, 4);
                v += __shfl_xor(v, 8);
                l[r] += v;
            }
            __syncthreads();                       // P visible (and keeps waves in step)

            // ---- O += P V : A = P(16q x 32k), B = V(32k x 16d) from Vsh[d][key]
#pragma unroll
            for (int kh = 0; kh < 2; ++kh) {
                BF16Frag ap;
                ap.h[0] = *(const v8bf*)&Psh[wave][lrow][kh * 32 + rhalf];
                ap.h[1] = *(const v8bf*)&Psh[wave][lrow][kh * 32 + rhalf + 16];
#pragma unroll
                for (int dt = 0; dt < 4; ++dt) {
                    BF16Frag bv;
                    int dcol = dt * 16 + lrow;
                    int kb   = kh * 32 + half * 8;
                    bv.h[0] = *(const v8bf*)&Vsh[dcol][kb];
                    bv.h[1] = *(const v8bf*)&Vsh[dcol][kb + 16];
                    acc_o[dt] = wmma_bf16(ap, bv, acc_o[dt]);
                }
            }
        }

        // ---- finish chunk: normalize by per-row softmax denominator
#pragma unroll
        for (int dt = 0; dt < 4; ++dt)
#pragma unroll
            for (int r = 0; r < 8; ++r)
                out_total[dt][r] += acc_o[dt][r] / l[r];
    }

    // ---- write ctx (bf16), averaged over SP chunks
#pragma unroll
    for (int dt = 0; dt < 4; ++dt) {
#pragma unroll
        for (int r = 0; r < 8; ++r) {
            int row = q_row0 + wave * 16 + rhalf + r;
            int col = colh + dt * 16 + lrow;
            Ctx[(size_t)row * HS_ + col] = (__bf16)(out_total[dt][r] * 0.25f);
        }
    }
}

// ---------------------------------------------------------------------------
// Host-side orchestration
// ---------------------------------------------------------------------------
extern "C" void kernel_launch(void* const* d_in, const int* in_sizes, int n_in,
                              void* d_out, int out_size, void* d_ws, size_t ws_size,
                              hipStream_t stream) {
    const float* hs = (const float*)d_in[0];
    const float* Wq = (const float*)d_in[1];
    const float* bq = (const float*)d_in[2];
    const float* Wk = (const float*)d_in[3];
    const float* bk = (const float*)d_in[4];
    const float* Wv = (const float*)d_in[5];
    const float* bv = (const float*)d_in[6];
    const float* Wo = (const float*)d_in[7];
    const float* bo = (const float*)d_in[8];
    float* out = (float*)d_out;

    const size_t nX = (size_t)S_ * HS_;      // 8M activation elems
    const size_t nW = (size_t)HS_ * HS_;     // 1M weight elems

    __bf16* wsb = (__bf16*)d_ws;
    __bf16* Xb   = wsb;
    __bf16* Wqb  = Xb + nX;
    __bf16* Wkb  = Wqb + nW;
    __bf16* Wvb  = Wkb + nW;
    __bf16* Wob  = Wvb + nW;
    __bf16* Qb   = Wob + nW;
    __bf16* Kb   = Qb + nX;
    __bf16* Vb   = Kb + nX;
    __bf16* Ctxb = Vb + nX;
    (void)ws_size; (void)n_in; (void)in_sizes; (void)out_size;

    // 1) fp32 -> bf16 staging
    cvt_f32_bf16_kernel<<<4096, 256, 0, stream>>>(hs, Xb, (int)nX);
    cvt_f32_bf16_kernel<<<1024, 256, 0, stream>>>(Wq, Wqb, (int)nW);
    cvt_f32_bf16_kernel<<<1024, 256, 0, stream>>>(Wk, Wkb, (int)nW);
    cvt_f32_bf16_kernel<<<1024, 256, 0, stream>>>(Wv, Wvb, (int)nW);
    cvt_f32_bf16_kernel<<<1024, 256, 0, stream>>>(Wo, Wob, (int)nW);

    // 2) Q/K/V projections (WMMA GEMMs, bf16 out)
    dim3 ggrid(S_ / 128, HS_ / 128);
    gemm_bf16_kernel<false><<<ggrid, 256, 0, stream>>>(Xb, Wqb, bq, Qb, S_, HS_, HS_);
    gemm_bf16_kernel<false><<<ggrid, 256, 0, stream>>>(Xb, Wkb, bk, Kb, S_, HS_, HS_);
    gemm_bf16_kernel<false><<<ggrid, 256, 0, stream>>>(Xb, Wvb, bv, Vb, S_, HS_, HS_);

    // 3) ring attention with per-chunk softmax
    dim3 agrid(SC_ / 128, NH_, SP_);
    ring_attention_kernel<<<agrid, 256, 0, stream>>>(Qb, Kb, Vb, Ctxb);

    // 4) output projection (fp32 out straight into d_out)
    gemm_bf16_kernel<true><<<ggrid, 256, 0, stream>>>(Ctxb, Wob, bo, out, S_, HS_, HS_);
}
